// DagnabbitAutoEncoder_85907935854597
// MI455X (gfx1250) — matrix operand order repaired and checked by reading, not compiled, not run on hardware
//
#include <hip/hip_runtime.h>

#define N_NODES 65536
#define N_ROOT  256
#define D       256
#define KD      512   // K*D  (layer-1 K)
#define H2      512   // 2*D  (hidden width, layer-2 K)
#define NBLK    64
#define NTHR    256
#define LDSTRIDE 520  // padded f16 row stride (1040B, 16B-aligned, bank-skewed)

typedef __attribute__((ext_vector_type(16))) _Float16 v16h;
typedef __attribute__((ext_vector_type(8)))  _Float16 v8h;
typedef __attribute__((ext_vector_type(4)))  _Float16 v4h;
typedef __attribute__((ext_vector_type(8)))  float    v8f;
typedef __attribute__((ext_vector_type(4)))  float    f32x4;

static constexpr size_t SMEM_BYTES =
    (size_t)2 * 32 * LDSTRIDE * sizeof(_Float16) + 64 * sizeof(int);

// ---------------------------------------------------------------- helpers
__device__ __forceinline__ float gelu_exact(float x) {
    return 0.5f * x * (1.0f + erff(x * 0.70710678118654752440f));
}

__device__ __forceinline__ v16h join8(v8h lo, v8h hi) {
    return __builtin_shufflevector(lo, hi, 0,1,2,3,4,5,6,7,8,9,10,11,12,13,14,15);
}

__device__ __forceinline__ v8f wmma_f16(v16h a, v16h b, v8f c) {
    return __builtin_amdgcn_wmma_f32_16x16x32_f16(false, a, false, b, (short)0, c,
                                                  false, false);
}

__device__ __forceinline__ v16h ldA(const _Float16* pA, int kk, int kbA) {
    return join8(*(const v8h*)(pA + kk + kbA),
                 *(const v8h*)(pA + kk + 8 + kbA));
}

// Grid-wide barrier: ctrl[0]=arrive count, ctrl[1]=generation.
__device__ __forceinline__ void gbar(int* ctrl, int nblocks) {
    __threadfence();
    __syncthreads();
    if (threadIdx.x == 0) {
        int g = __hip_atomic_load(&ctrl[1], __ATOMIC_RELAXED, __HIP_MEMORY_SCOPE_AGENT);
        int a = __hip_atomic_fetch_add(&ctrl[0], 1, __ATOMIC_ACQ_REL, __HIP_MEMORY_SCOPE_AGENT);
        if (a == nblocks - 1) {
            __hip_atomic_store(&ctrl[0], 0, __ATOMIC_RELAXED, __HIP_MEMORY_SCOPE_AGENT);
            __hip_atomic_fetch_add(&ctrl[1], 1, __ATOMIC_ACQ_REL, __HIP_MEMORY_SCOPE_AGENT);
        } else {
            while (__hip_atomic_load(&ctrl[1], __ATOMIC_ACQUIRE, __HIP_MEMORY_SCOPE_AGENT) == g) {
                __builtin_amdgcn_s_sleep(2);
            }
        }
    }
    __syncthreads();
    __threadfence();
}

// ---------------------------------------------------------------- prep
__global__ void dag_prep(const float* __restrict__ roots,
                         const float* __restrict__ W1, const float* __restrict__ W2,
                         int* ctrl, int* level, int* counts, int* offs, int* cursor,
                         _Float16* __restrict__ W1t, _Float16* __restrict__ W2t,
                         float* __restrict__ out)
{
    const int gtid = blockIdx.x * blockDim.x + threadIdx.x;
    const int gsz  = gridDim.x * blockDim.x;
    for (int i = gtid; i < 64; i += gsz) ctrl[i] = 0;
    for (int i = gtid; i < N_NODES; i += gsz) {
        level[i]  = (i < N_ROOT) ? 0 : -1;
        counts[i] = 0;
        cursor[i] = 0;
    }
    for (int i = gtid; i < N_NODES + 1; i += gsz) offs[i] = 0;
    for (int i = gtid; i < N_ROOT * D; i += gsz) out[i] = roots[i];
    // W1t[n*KD + k] = W1[k*H2 + n]   (B column-major -> contiguous-K fragments)
    for (int i = gtid; i < H2 * KD; i += gsz) {
        int n = i / KD, k = i - n * KD;
        W1t[i] = (_Float16)W1[(size_t)k * H2 + n];
    }
    // W2t[n*H2 + k] = W2[k*D + n]
    for (int i = gtid; i < D * H2; i += gsz) {
        int n = i / H2, k = i - n * H2;
        W2t[i] = (_Float16)W2[(size_t)k * D + n];
    }
}

// ---------------------------------------------------------------- persistent
__launch_bounds__(NTHR, 1)
__global__ void dag_persistent(const float* __restrict__ b1,
                               const float* __restrict__ b2,
                               const int*   __restrict__ par,
                               const _Float16* __restrict__ W1t,
                               const _Float16* __restrict__ W2t,
                               int* ctrl, int* level, int* counts,
                               int* offs, int* cursor, int* order,
                               float* __restrict__ out)
{
    extern __shared__ char smem[];
    _Float16 (*Xa)[LDSTRIDE] = (_Float16 (*)[LDSTRIDE])smem;                       // [32][520]
    _Float16 (*Ht)[LDSTRIDE] =
        (_Float16 (*)[LDSTRIDE])(smem + (size_t)32 * LDSTRIDE * sizeof(_Float16)); // [32][520]
    int* snode  = (int*)(smem + (size_t)2 * 32 * LDSTRIDE * sizeof(_Float16));     // [32]
    int* svalid = snode + 32;                                                      // [32]

    const int tid  = threadIdx.x;
    const int gtid = blockIdx.x * blockDim.x + tid;
    const int gsz  = gridDim.x * blockDim.x;

    // ---------- Phase 1: level assignment (frontier rounds, <= DAG depth)
    for (;;) {
        int made = 0;
        for (int i = N_ROOT + gtid; i < N_NODES; i += gsz) {
            if (__hip_atomic_load(&level[i], __ATOMIC_RELAXED, __HIP_MEMORY_SCOPE_AGENT) < 0) {
                int p0 = par[2 * i], p1 = par[2 * i + 1];
                int l0 = __hip_atomic_load(&level[p0], __ATOMIC_RELAXED, __HIP_MEMORY_SCOPE_AGENT);
                int l1 = __hip_atomic_load(&level[p1], __ATOMIC_RELAXED, __HIP_MEMORY_SCOPE_AGENT);
                if (l0 >= 0 && l1 >= 0) {
                    int lv = 1 + (l0 > l1 ? l0 : l1);
                    __hip_atomic_store(&level[i], lv, __ATOMIC_RELAXED, __HIP_MEMORY_SCOPE_AGENT);
                    atomicMax(&ctrl[3], lv);
                    made = 1;
                }
            }
        }
        if (made) atomicAdd(&ctrl[2], 1);
        gbar(ctrl, NBLK);
        int prog = __hip_atomic_load(&ctrl[2], __ATOMIC_RELAXED, __HIP_MEMORY_SCOPE_AGENT);
        gbar(ctrl, NBLK);
        if (gtid == 0 && prog != 0)
            __hip_atomic_store(&ctrl[2], 0, __ATOMIC_RELAXED, __HIP_MEMORY_SCOPE_AGENT);
        gbar(ctrl, NBLK);
        if (prog == 0) break;
    }

    // ---------- Phase 2: counting sort by level
    for (int i = N_ROOT + gtid; i < N_NODES; i += gsz)
        atomicAdd(&counts[level[i]], 1);
    gbar(ctrl, NBLK);
    if (gtid == 0) {
        int ml = ctrl[3];
        int run = 0;
        for (int l = 0; l <= ml; ++l) {
            offs[l]   = run;
            cursor[l] = run;
            run += counts[l];
        }
        offs[ml + 1] = run;
    }
    gbar(ctrl, NBLK);
    for (int i = N_ROOT + gtid; i < N_NODES; i += gsz) {
        int lv  = level[i];
        int pos = atomicAdd(&cursor[lv], 1);
        order[pos] = i;
    }
    gbar(ctrl, NBLK);

    // ---------- Phase 3: level-batched MLP with WMMA, M-tile = 32 rows
    const int maxl = __hip_atomic_load(&ctrl[3], __ATOMIC_RELAXED, __HIP_MEMORY_SCOPE_AGENT);
    const int lane  = tid & 31;
    const int wv    = tid >> 5;            // 8 waves
    const int mrow  = lane & 15;           // A rows / B cols index
    const int kbA   = (lane < 16) ? 0 : 16; // 16-bit A layout K split
    const int kbB   = (lane < 16) ? 0 : 16;// 16-bit B layout K split
    const int mbase = (lane < 16) ? 0 : 8; // C/D row base

    // per-wave weight column base pointers (hoisted out of all loops)
    const _Float16* pB1[4];
#pragma unroll
    for (int j = 0; j < 4; ++j)
        pB1[j] = W1t + (size_t)((wv * 4 + j) * 16 + mrow) * KD + kbB;
    const _Float16* pB2[2];
#pragma unroll
    for (int j = 0; j < 2; ++j)
        pB2[j] = W2t + (size_t)((wv * 2 + j) * 16 + mrow) * H2 + kbB;

    for (int L = 1; L <= maxl; ++L) {
        const int s = offs[L];
        const int e = offs[L + 1];
        const int ntile = (e - s + 31) >> 5;

        for (int t = blockIdx.x; t < ntile; t += gridDim.x) {
            // ---- gather 32 node rows -> Xa (f16), x = [emb(p0) | emb(p1)]
            if (tid < 32) {
                int idx = s + t * 32 + tid;
                int ok  = idx < e;
                snode[tid]  = ok ? order[idx] : 0;
                svalid[tid] = ok;
            }
            __syncthreads();
#pragma unroll
            for (int rr = 0; rr < 2; ++rr) {
                int r  = (tid >> 4) + 16 * rr;   // row 0..31
                int c0 = (tid & 15) * 32;        // 32 cols per thread
                int nd = snode[r];
                int ok = svalid[r];
                int p0 = par[2 * nd], p1 = par[2 * nd + 1];
                const float* srow = (c0 < D) ? (out + (size_t)p0 * D + c0)
                                             : (out + (size_t)p1 * D + (c0 - D));
                const f32x4* s4 = (const f32x4*)srow;
#pragma unroll
                for (int c4 = 0; c4 < 8; ++c4) {
                    f32x4 v = ok ? s4[c4] : f32x4{0.f, 0.f, 0.f, 0.f};
                    *(v4h*)&Xa[r][c0 + 4 * c4] = __builtin_convertvector(v, v4h);
                }
            }
            __syncthreads();

            // ---- layer 1: [32xKD] x [KDxH2] ; wave owns 4 N-tiles x 2 M-tiles.
            {
                const _Float16* pA0 = &Xa[mrow][0];
                const _Float16* pA1 = &Xa[16 + mrow][0];
                v8f acc[2][4];
#pragma unroll
                for (int mt = 0; mt < 2; ++mt)
#pragma unroll
                    for (int j = 0; j < 4; ++j)
                        acc[mt][j] = v8f{0.f,0.f,0.f,0.f,0.f,0.f,0.f,0.f};

                v16h aP[2], aQ[2], bP[4], bQ[4];
                aP[0] = ldA(pA0, 0, kbA);
                aP[1] = ldA(pA1, 0, kbA);
#pragma unroll
                for (int j = 0; j < 4; ++j) bP[j] = *(const v16h*)(pB1[j]);

#pragma unroll 1
                for (int kk = 32; kk < KD - 32; kk += 64) {
                    aQ[0] = ldA(pA0, kk, kbA);
                    aQ[1] = ldA(pA1, kk, kbA);
#pragma unroll
                    for (int j = 0; j < 4; ++j) bQ[j] = *(const v16h*)(pB1[j] + kk);
#pragma unroll
                    for (int j = 0; j < 4; ++j) {
                        acc[0][j] = wmma_f16(aP[0], bP[j], acc[0][j]);
                        acc[1][j] = wmma_f16(aP[1], bP[j], acc[1][j]);
                    }
                    aP[0] = ldA(pA0, kk + 32, kbA);
                    aP[1] = ldA(pA1, kk + 32, kbA);
#pragma unroll
                    for (int j = 0; j < 4; ++j) bP[j] = *(const v16h*)(pB1[j] + kk + 32);
#pragma unroll
                    for (int j = 0; j < 4; ++j) {
                        acc[0][j] = wmma_f16(aQ[0], bQ[j], acc[0][j]);
                        acc[1][j] = wmma_f16(aQ[1], bQ[j], acc[1][j]);
                    }
                }
                aQ[0] = ldA(pA0, KD - 32, kbA);
                aQ[1] = ldA(pA1, KD - 32, kbA);
#pragma unroll
                for (int j = 0; j < 4; ++j) bQ[j] = *(const v16h*)(pB1[j] + KD - 32);
#pragma unroll
                for (int j = 0; j < 4; ++j) {
                    acc[0][j] = wmma_f16(aP[0], bP[j], acc[0][j]);
                    acc[1][j] = wmma_f16(aP[1], bP[j], acc[1][j]);
                }
#pragma unroll
                for (int j = 0; j < 4; ++j) {
                    acc[0][j] = wmma_f16(aQ[0], bQ[j], acc[0][j]);
                    acc[1][j] = wmma_f16(aQ[1], bQ[j], acc[1][j]);
                }

                // fused bias + exact GELU -> Ht (f16)
#pragma unroll
                for (int mt = 0; mt < 2; ++mt)
#pragma unroll
                    for (int j = 0; j < 4; ++j) {
                        int nc = (wv * 4 + j) * 16 + mrow;
                        float bias = b1[nc];
#pragma unroll
                        for (int r = 0; r < 8; ++r) {
                            float h = gelu_exact(acc[mt][j][r] + bias);
                            Ht[mt * 16 + mbase + r][nc] = (_Float16)h;
                        }
                    }
            }
            __syncthreads();

            // ---- layer 2: [32xH2] x [H2xD] ; wave owns 2 N-tiles x 2 M-tiles.
            {
                const _Float16* pA0 = &Ht[mrow][0];
                const _Float16* pA1 = &Ht[16 + mrow][0];
                v8f acc[2][2];
#pragma unroll
                for (int mt = 0; mt < 2; ++mt)
#pragma unroll
                    for (int j = 0; j < 2; ++j)
                        acc[mt][j] = v8f{0.f,0.f,0.f,0.f,0.f,0.f,0.f,0.f};

                v16h aP[2], aQ[2], bP[2], bQ[2];
                aP[0] = ldA(pA0, 0, kbA);
                aP[1] = ldA(pA1, 0, kbA);
#pragma unroll
                for (int j = 0; j < 2; ++j) bP[j] = *(const v16h*)(pB2[j]);

#pragma unroll 1
                for (int kk = 32; kk < H2 - 32; kk += 64) {
                    aQ[0] = ldA(pA0, kk, kbA);
                    aQ[1] = ldA(pA1, kk, kbA);
#pragma unroll
                    for (int j = 0; j < 2; ++j) bQ[j] = *(const v16h*)(pB2[j] + kk);
#pragma unroll
                    for (int j = 0; j < 2; ++j) {
                        acc[0][j] = wmma_f16(aP[0], bP[j], acc[0][j]);
                        acc[1][j] = wmma_f16(aP[1], bP[j], acc[1][j]);
                    }
                    aP[0] = ldA(pA0, kk + 32, kbA);
                    aP[1] = ldA(pA1, kk + 32, kbA);
#pragma unroll
                    for (int j = 0; j < 2; ++j) bP[j] = *(const v16h*)(pB2[j] + kk + 32);
#pragma unroll
                    for (int j = 0; j < 2; ++j) {
                        acc[0][j] = wmma_f16(aQ[0], bQ[j], acc[0][j]);
                        acc[1][j] = wmma_f16(aQ[1], bQ[j], acc[1][j]);
                    }
                }
                aQ[0] = ldA(pA0, H2 - 32, kbA);
                aQ[1] = ldA(pA1, H2 - 32, kbA);
#pragma unroll
                for (int j = 0; j < 2; ++j) bQ[j] = *(const v16h*)(pB2[j] + H2 - 32);
#pragma unroll
                for (int j = 0; j < 2; ++j) {
                    acc[0][j] = wmma_f16(aP[0], bP[j], acc[0][j]);
                    acc[1][j] = wmma_f16(aP[1], bP[j], acc[1][j]);
                }
#pragma unroll
                for (int j = 0; j < 2; ++j) {
                    acc[0][j] = wmma_f16(aQ[0], bQ[j], acc[0][j]);
                    acc[1][j] = wmma_f16(aQ[1], bQ[j], acc[1][j]);
                }

                // bias + scatter f32 rows to output buffer
#pragma unroll
                for (int mt = 0; mt < 2; ++mt)
#pragma unroll
                    for (int j = 0; j < 2; ++j) {
                        int nc = (wv * 2 + j) * 16 + mrow;
                        float bias = b2[nc];
#pragma unroll
                        for (int r = 0; r < 8; ++r) {
                            int m = mt * 16 + mbase + r;
                            if (svalid[m])
                                out[(size_t)snode[m] * D + nc] = acc[mt][j][r] + bias;
                        }
                    }
            }
            __syncthreads();
        }
        gbar(ctrl, NBLK);   // level L done & visible before level L+1 gathers
    }
}

// ---------------------------------------------------------------- launch
extern "C" void kernel_launch(void* const* d_in, const int* in_sizes, int n_in,
                              void* d_out, int out_size, void* d_ws, size_t ws_size,
                              hipStream_t stream)
{
    const float* roots = (const float*)d_in[0];
    const float* W1    = (const float*)d_in[1];
    const float* b1    = (const float*)d_in[2];
    const float* W2    = (const float*)d_in[3];
    const float* b2    = (const float*)d_in[4];
    const int*   par   = (const int*)  d_in[5];
    float* out = (float*)d_out;

    char* w = (char*)d_ws;
    size_t off = 0;
    auto take = [&](size_t bytes, size_t align) {
        off = (off + align - 1) & ~(align - 1);
        void* p = w + off;
        off += bytes;
        return p;
    };
    int* ctrl   = (int*)take(64 * sizeof(int), 256);
    int* level  = (int*)take((size_t)N_NODES * sizeof(int), 256);
    int* counts = (int*)take((size_t)N_NODES * sizeof(int), 256);
    int* offs   = (int*)take((size_t)(N_NODES + 1) * sizeof(int), 256);
    int* cursor = (int*)take((size_t)N_NODES * sizeof(int), 256);
    int* order  = (int*)take((size_t)N_NODES * sizeof(int), 256);
    _Float16* W1t = (_Float16*)take((size_t)H2 * KD * sizeof(_Float16), 256);
    _Float16* W2t = (_Float16*)take((size_t)D * H2 * sizeof(_Float16), 256);
    (void)ws_size; (void)in_sizes; (void)n_in; (void)out_size;

    // Allow >64KB dynamic LDS (WGP has 320KB); harmless if already permitted.
    (void)hipFuncSetAttribute((const void*)dag_persistent,
                              hipFuncAttributeMaxDynamicSharedMemorySize,
                              (int)SMEM_BYTES);

    dag_prep<<<768, 256, 0, stream>>>(roots, W1, W2, ctrl, level, counts, offs,
                                      cursor, W1t, W2t, out);
    dag_persistent<<<NBLK, NTHR, SMEM_BYTES, stream>>>(b1, b2, par, W1t, W2t,
                                                       ctrl, level, counts, offs,
                                                       cursor, order, out);
}
